// GdmGNN_30124900614315
// MI455X (gfx1250) — compile-verified, matrix operand order checked by reading, not run on hardware
//
#include <hip/hip_runtime.h>

typedef __attribute__((ext_vector_type(16))) __bf16         v16bf;
typedef __attribute__((ext_vector_type(8)))  float          v8f;
typedef __attribute__((ext_vector_type(8)))  unsigned short v8u16;
typedef unsigned short u16;

__device__ __forceinline__ float bf2f(u16 v) {
    union { unsigned int u; float f; } c; c.u = ((unsigned int)v) << 16; return c.f;
}
__device__ __forceinline__ u16 f2bf(float f) {
    union { float f; unsigned int u; } c; c.f = f;
    unsigned int r = (c.u + 0x7FFFu + ((c.u >> 16) & 1u)) >> 16;
    return (u16)r;
}

// One WMMA 16-bit operand fragment (ISA 7.12.2): lane&15 = row/col,
// K-phase = (lane>>4)*8; e0..e7 = K base..base+7, e8..e15 = K base+16..base+23.
__device__ __forceinline__ v16bf load_frag(const u16* p) {
    union { v8u16 h[2]; v16bf v; } u;
    u.h[0] = *(const v8u16*)(p);
    u.h[1] = *(const v8u16*)(p + 16);
    return u.v;
}
__device__ __forceinline__ v8f wmma_bf16(v16bf a, v16bf b, v8f c) {
    return __builtin_amdgcn_wmma_f32_16x16x32_bf16(false, a, false, b, (short)0, c, false, false);
}

// ---------------------------------------------------------------------------
// Batched bf16 WMMA GEMM, 32x32 register tile per wave (4 accumulators).
//   D[b] = A1[b] @ B1t[b]^T (+ A2 @ B2t^T if DUAL) + bias, optional relu.
//   A is MxK row-major (lda=K); Bt is NxK row-major (ldb=K).
// Single-pair path is software-pipelined (next-k frags loaded before WMMAs).
// ---------------------------------------------------------------------------
template<bool DUAL>
__global__ void __launch_bounds__(256)
wmma_gemm32(const u16* __restrict__ A, long strideA,
            const u16* __restrict__ Bt, long strideB,
            const u16* __restrict__ A2, const u16* __restrict__ B2t,
            const float* __restrict__ bias,
            float* __restrict__ outF, u16* __restrict__ outB, u16* __restrict__ outBT,
            int Mb, int N, int K, int relu)
{
    const int lane = threadIdx.x & 31;
    const int wid  = threadIdx.x >> 5;
    const int b    = blockIdx.z;
    const int tN   = N >> 5;
    const int tM   = Mb >> 5;
    const int tile = blockIdx.x * 8 + wid;
    if (tile >= tM * tN) return;
    const int tm = tile / tN, tn = tile % tN;

    const int ln15 = lane & 15;
    const int koff = (lane >> 4) * 8;

    const u16* ap0 = A  + (long)b * strideA + (long)(tm * 32 + ln15) * K + koff;
    const u16* ap1 = ap0 + 16l * K;
    const u16* bp0 = Bt + (long)b * strideB + (long)(tn * 32 + ln15) * K + koff;
    const u16* bp1 = bp0 + 16l * K;

    v8f c[2][2] = {};

    if (!DUAL) {
        v16bf a0 = load_frag(ap0), a1 = load_frag(ap1);
        v16bf b0 = load_frag(bp0), b1 = load_frag(bp1);
        int k = 0;
        for (; k + 32 < K; k += 32) {
            __builtin_prefetch(ap0 + k + 96, 0, 1);
            __builtin_prefetch(ap1 + k + 96, 0, 1);
            v16bf na0 = load_frag(ap0 + k + 32);
            v16bf na1 = load_frag(ap1 + k + 32);
            v16bf nb0 = load_frag(bp0 + k + 32);
            v16bf nb1 = load_frag(bp1 + k + 32);
            c[0][0] = wmma_bf16(a0, b0, c[0][0]);
            c[0][1] = wmma_bf16(a0, b1, c[0][1]);
            c[1][0] = wmma_bf16(a1, b0, c[1][0]);
            c[1][1] = wmma_bf16(a1, b1, c[1][1]);
            a0 = na0; a1 = na1; b0 = nb0; b1 = nb1;
        }
        c[0][0] = wmma_bf16(a0, b0, c[0][0]);
        c[0][1] = wmma_bf16(a0, b1, c[0][1]);
        c[1][0] = wmma_bf16(a1, b0, c[1][0]);
        c[1][1] = wmma_bf16(a1, b1, c[1][1]);
    } else {
        const u16* ap20 = A2  + (long)(tm * 32 + ln15) * K + koff;
        const u16* ap21 = ap20 + 16l * K;
        const u16* bp20 = B2t + (long)(tn * 32 + ln15) * K + koff;
        const u16* bp21 = bp20 + 16l * K;
        for (int k = 0; k < K; k += 32) {
            v16bf a0 = load_frag(ap0 + k), a1 = load_frag(ap1 + k);
            v16bf b0 = load_frag(bp0 + k), b1 = load_frag(bp1 + k);
            c[0][0] = wmma_bf16(a0, b0, c[0][0]);
            c[0][1] = wmma_bf16(a0, b1, c[0][1]);
            c[1][0] = wmma_bf16(a1, b0, c[1][0]);
            c[1][1] = wmma_bf16(a1, b1, c[1][1]);
            v16bf d0 = load_frag(ap20 + k), d1 = load_frag(ap21 + k);
            v16bf e0 = load_frag(bp20 + k), e1 = load_frag(bp21 + k);
            c[0][0] = wmma_bf16(d0, e0, c[0][0]);
            c[0][1] = wmma_bf16(d0, e1, c[0][1]);
            c[1][0] = wmma_bf16(d1, e0, c[1][0]);
            c[1][1] = wmma_bf16(d1, e1, c[1][1]);
        }
    }

    const int  half = (lane >> 4) * 8;
    float bv[2];
    bv[0] = bias ? bias[tn * 32 + ln15]      : 0.0f;
    bv[1] = bias ? bias[tn * 32 + 16 + ln15] : 0.0f;
    #pragma unroll
    for (int ih = 0; ih < 2; ++ih) {
        #pragma unroll
        for (int jh = 0; jh < 2; ++jh) {
            const int n = tn * 32 + jh * 16 + ln15;
            #pragma unroll
            for (int e = 0; e < 8; ++e) {
                float v = c[ih][jh][e] + bv[jh];
                if (relu) v = fmaxf(v, 0.0f);
                const long r = (long)b * Mb + tm * 32 + ih * 16 + e + half;
                if (outF) outF[r * N + n] = v;
                if (outB || outBT) {
                    const u16 bw = f2bf(v);
                    if (outB)  outB[r * N + n] = bw;
                    if (outBT) outBT[((r >> 9) * (long)N + n) * 512 + (r & 511)] = bw;
                }
            }
        }
    }
}

// ---------------------------------------------------------------------------
// Fused Gram + Gaussian adjacency, 32x32 tile per wave, upper-triangular
// tiles only (A is symmetric; mirror-store the bf16 result to [j,i]).
// grid: 32 blocks (256 tiles, lower-tri waves exit) x 64 windows.
// ---------------------------------------------------------------------------
__global__ void __launch_bounds__(256)
gram_adj_bf16(const u16* __restrict__ h, const float* __restrict__ sq,
              const float* __restrict__ sigma, u16* __restrict__ Aout, int F)
{
    const int lane = threadIdx.x & 31;
    const int wid  = threadIdx.x >> 5;
    const int b    = blockIdx.z;
    const int tile = blockIdx.x * 8 + wid;     // 0..255 (16x16 tiles of 32x32)
    const int ti = tile >> 4, tj = tile & 15;
    if (tj < ti) return;                       // symmetric: upper triangle only

    const u16* hb  = h + (long)b * 512 * F;
    const int ln15 = lane & 15;
    const int koff = (lane >> 4) * 8;
    const u16* ap0 = hb + (long)(ti * 32 + ln15) * F + koff;
    const u16* ap1 = ap0 + 16l * F;
    const u16* bp0 = hb + (long)(tj * 32 + ln15) * F + koff;
    const u16* bp1 = bp0 + 16l * F;

    v8f c[2][2] = {};
    v16bf a0 = load_frag(ap0), a1 = load_frag(ap1);
    v16bf b0 = load_frag(bp0), b1 = load_frag(bp1);
    int k = 0;
    for (; k + 32 < F; k += 32) {
        v16bf na0 = load_frag(ap0 + k + 32);
        v16bf na1 = load_frag(ap1 + k + 32);
        v16bf nb0 = load_frag(bp0 + k + 32);
        v16bf nb1 = load_frag(bp1 + k + 32);
        c[0][0] = wmma_bf16(a0, b0, c[0][0]);
        c[0][1] = wmma_bf16(a0, b1, c[0][1]);
        c[1][0] = wmma_bf16(a1, b0, c[1][0]);
        c[1][1] = wmma_bf16(a1, b1, c[1][1]);
        a0 = na0; a1 = na1; b0 = nb0; b1 = nb1;
    }
    c[0][0] = wmma_bf16(a0, b0, c[0][0]);
    c[0][1] = wmma_bf16(a0, b1, c[0][1]);
    c[1][0] = wmma_bf16(a1, b0, c[1][0]);
    c[1][1] = wmma_bf16(a1, b1, c[1][1]);

    const float s    = sigma[0];
    const float coef = -1.0f / (2.0f * s * s);
    const int   half = (lane >> 4) * 8;
    float sj[2];
    sj[0] = sq[b * 512 + tj * 32 + ln15];
    sj[1] = sq[b * 512 + tj * 32 + 16 + ln15];
    #pragma unroll
    for (int ih = 0; ih < 2; ++ih) {
        #pragma unroll
        for (int e = 0; e < 8; ++e) {
            const int   mi = ti * 32 + ih * 16 + e + half;
            const float si = sq[b * 512 + mi];
            #pragma unroll
            for (int jh = 0; jh < 2; ++jh) {
                const int nj = tj * 32 + jh * 16 + ln15;
                float d = fmaxf(si + sj[jh] - 2.0f * c[ih][jh][e], 0.0f);
                float a = __expf(d * coef);
                if (mi == nj) a = 0.0f;
                const u16 bw = f2bf(a);
                Aout[((long)b * 512 + mi) * 512 + nj] = bw;
                if (ti != tj) Aout[((long)b * 512 + nj) * 512 + mi] = bw;
            }
        }
    }
}

// --------------------------- small helper kernels --------------------------
__global__ void cvt_f32_bf16(const float* __restrict__ in, u16* __restrict__ out, int n) {
    int i = blockIdx.x * blockDim.x + threadIdx.x;
    if (i < n) out[i] = f2bf(in[i]);
}
// W[KxN] f32 -> WT[NxK] bf16 (WMMA B-operand layout)
__global__ void cvt_transpose_bf16(const float* __restrict__ in, u16* __restrict__ outT,
                                   int K, int N) {
    int i = blockIdx.x * blockDim.x + threadIdx.x;
    if (i < K * N) { int k = i / N, n = i % N; outT[n * K + k] = f2bf(in[i]); }
}
__global__ void rownorm_sq(const u16* __restrict__ h, float* __restrict__ sq, int F) {
    int r = blockIdx.x * blockDim.x + threadIdx.x;   // one thread per row
    const u16* p = h + (long)r * F;
    float s = 0.f;
    for (int k = 0; k < F; k += 8) {
        v8u16 v = *(const v8u16*)(p + k);
        #pragma unroll
        for (int e = 0; e < 8; ++e) { float f = bf2f(v[e]); s += f * f; }
    }
    sq[r] = s;
}
__global__ void pool_mean(const float* __restrict__ h3, float* __restrict__ g) {
    int b = blockIdx.x, f = threadIdx.x;             // 64 x 256
    const float* p = h3 + (long)b * 512 * 256 + f;
    float s = 0.f;
    for (int n = 0; n < 512; ++n) s += p[n * 256];
    g[b * 256 + f] = s * (1.0f / 512.0f);
}
__global__ void mlp_head(const float* __restrict__ g, const float* __restrict__ W1,
                         const float* __restrict__ b1, const float* __restrict__ Wout,
                         const float* __restrict__ bout, float* __restrict__ out) {
    __shared__ float t[256];
    int b = blockIdx.x, j = threadIdx.x;
    const float* gb = g + b * 256;
    float s = b1[j];
    for (int k = 0; k < 256; ++k) s += gb[k] * W1[k * 256 + j];
    t[j] = fmaxf(s, 0.0f);
    __syncthreads();
    if (j < 6) {
        float o = bout[j];
        for (int k = 0; k < 256; ++k) o += t[k] * Wout[k * 6 + j];
        out[b * 6 + j] = o;
    }
}

// ---------------------------------------------------------------------------
extern "C" void kernel_launch(void* const* d_in, const int* in_sizes, int n_in,
                              void* d_out, int out_size, void* d_ws, size_t ws_size,
                              hipStream_t stream) {
    const float* x      = (const float*)d_in[0];
    const float* sigma  = (const float*)d_in[4];
    const float* Wg     = (const float*)d_in[5];
    const float* bg     = (const float*)d_in[6];
    const float* Wrel0  = (const float*)d_in[7];
    const float* Wroot0 = (const float*)d_in[8];
    const float* brel0  = (const float*)d_in[9];
    const float* Wrel1  = (const float*)d_in[10];
    const float* Wroot1 = (const float*)d_in[11];
    const float* brel1  = (const float*)d_in[12];
    const float* Wrel2  = (const float*)d_in[13];
    const float* Wroot2 = (const float*)d_in[14];
    const float* brel2  = (const float*)d_in[15];
    const float* W1     = (const float*)d_in[16];
    const float* b1     = (const float*)d_in[17];
    const float* Wout   = (const float*)d_in[18];
    const float* bout   = (const float*)d_in[19];
    float* out = (float*)d_out;

    char* ws = (char*)d_ws;
    const size_t MB = 1024ull * 1024ull;
    // workspace layout (peak ~154 MB, with aliasing):
    u16*   Aadj   = (u16*)(ws + 0);            // 32MB adjacency bf16 [64][512][512]
    float* h3     = (float*)(ws + 0);          // 32MB f32, reuses Aadj after last A@h
    u16*   xb     = (u16*)(ws + 32 * MB);      // 8MB x bf16
    u16*   agg0   = xb;                        // 8MB, reuses xb after GEMM1
    u16*   h0b    = (u16*)(ws + 40 * MB);      // 8MB
    u16*   h0bT   = (u16*)(ws + 48 * MB);      // 8MB [64][128][512]
    float* sqv    = (float*)(ws + 56 * MB);    // 128KB
    float* g      = (float*)(ws + 56 * MB);    // reuses sqv
    u16*   WgT    = (u16*)(ws + 57 * MB);
    u16*   Wrel0T = (u16*)(ws + 57 * MB +  64 * 1024);
    u16*   Wroot0T= (u16*)(ws + 57 * MB + 128 * 1024);
    u16*   Wrel1T = (u16*)(ws + 57 * MB + 192 * 1024);
    u16*   Wroot1T= (u16*)(ws + 57 * MB + 320 * 1024);
    u16*   Wrel2T = (u16*)(ws + 57 * MB + 448 * 1024);
    u16*   Wroot2T= (u16*)(ws + 57 * MB + 576 * 1024);
    u16*   h1b    = (u16*)(ws + 58 * MB);      // 16MB
    u16*   h1bT   = (u16*)(ws + 74 * MB);      // 16MB [64][256][512]
    u16*   agg1   = (u16*)(ws + 90 * MB);      // 16MB
    u16*   h2b    = (u16*)(ws + 106 * MB);     // 16MB
    u16*   h2bT   = (u16*)(ws + 122 * MB);     // 16MB
    u16*   agg2   = (u16*)(ws + 138 * MB);     // 16MB

    // --- precision conversion (weights transposed to NxK for the B operand) ---
    cvt_transpose_bf16<<<(128 * 128 + 255) / 256, 256, 0, stream>>>(Wg, WgT, 128, 128);
    cvt_transpose_bf16<<<(128 * 256 + 255) / 256, 256, 0, stream>>>(Wrel0, Wrel0T, 128, 256);
    cvt_transpose_bf16<<<(128 * 256 + 255) / 256, 256, 0, stream>>>(Wroot0, Wroot0T, 128, 256);
    cvt_transpose_bf16<<<(256 * 256 + 255) / 256, 256, 0, stream>>>(Wrel1, Wrel1T, 256, 256);
    cvt_transpose_bf16<<<(256 * 256 + 255) / 256, 256, 0, stream>>>(Wroot1, Wroot1T, 256, 256);
    cvt_transpose_bf16<<<(256 * 256 + 255) / 256, 256, 0, stream>>>(Wrel2, Wrel2T, 256, 256);
    cvt_transpose_bf16<<<(256 * 256 + 255) / 256, 256, 0, stream>>>(Wroot2, Wroot2T, 256, 256);
    cvt_f32_bf16<<<(32768 * 128) / 256, 256, 0, stream>>>(x, xb, 32768 * 128);

    // --- h0 = x @ Wg + bg  (M=32768, N=128, K=128): 1024x4 tiles -> 512 blocks ---
    wmma_gemm32<false><<<dim3(512, 1, 1), 256, 0, stream>>>(
        xb, 0, WgT, 0, nullptr, nullptr, bg, nullptr, h0b, h0bT, 32768, 128, 128, 0);

    // --- row squared norms, then fused Gram + Gaussian adjacency ---
    rownorm_sq<<<128, 256, 0, stream>>>(h0b, sqv, 128);
    gram_adj_bf16<<<dim3(32, 1, 64), 256, 0, stream>>>(h0b, sqv, sigma, Aadj, 128);

    // --- conv0: agg0 = A @ h0 ; h1 = relu(agg0@Wrel0 + h0@Wroot0 + brel0) ---
    wmma_gemm32<false><<<dim3(8, 1, 64), 256, 0, stream>>>(
        Aadj, 512l * 512, h0bT, 128l * 512, nullptr, nullptr, nullptr,
        nullptr, agg0, nullptr, 512, 128, 512, 0);
    wmma_gemm32<true><<<dim3(1024, 1, 1), 256, 0, stream>>>(
        agg0, 0, Wrel0T, 0, h0b, Wroot0T, brel0, nullptr, h1b, h1bT, 32768, 256, 128, 1);

    // --- conv1 ---
    wmma_gemm32<false><<<dim3(16, 1, 64), 256, 0, stream>>>(
        Aadj, 512l * 512, h1bT, 256l * 512, nullptr, nullptr, nullptr,
        nullptr, agg1, nullptr, 512, 256, 512, 0);
    wmma_gemm32<true><<<dim3(1024, 1, 1), 256, 0, stream>>>(
        agg1, 0, Wrel1T, 0, h1b, Wroot1T, brel1, nullptr, h2b, h2bT, 32768, 256, 256, 1);

    // --- conv2 (no relu, f32 out; h3 aliases Aadj which is now dead) ---
    wmma_gemm32<false><<<dim3(16, 1, 64), 256, 0, stream>>>(
        Aadj, 512l * 512, h2bT, 256l * 512, nullptr, nullptr, nullptr,
        nullptr, agg2, nullptr, 512, 256, 512, 0);
    wmma_gemm32<true><<<dim3(1024, 1, 1), 256, 0, stream>>>(
        agg2, 0, Wrel2T, 0, h2b, Wroot2T, brel2, h3, nullptr, nullptr, 32768, 256, 256, 0);

    // --- pooling + MLP head ---
    pool_mean<<<64, 256, 0, stream>>>(h3, g);
    mlp_head<<<64, 256, 0, stream>>>(g, W1, b1, Wout, bout, out);
}
